// TMSSM_81441169866921
// MI455X (gfx1250) — matrix operand order repaired and checked by reading, not compiled, not run on hardware
//
#include <hip/hip_runtime.h>
#include <hip/hip_bf16.h>
#include <math.h>

// ---------------------------------------------------------------------------
// Selective-SSM with DENSE 64x64 matrix recurrence, CDNA5 (gfx1250) wave32.
//   z_l = dA_l @ z_{l-1} + dBu_l      (64x64 matmul per step per batch)
// 3-pass chunked associative scan; matmuls on v_wmma_f32_16x16x32_bf16;
// chunk operand blocks staged into LDS with the Tensor Data Mover
// (tensor_load_to_lds / TENSORcnt) so the sequential loop never touches HBM.
// ---------------------------------------------------------------------------

typedef __bf16 bf16_t;
typedef __attribute__((ext_vector_type(16))) __bf16 v16bf;
typedef __attribute__((ext_vector_type(8)))  __bf16 v8bf;
typedef __attribute__((ext_vector_type(8)))  float  v8f;
typedef __attribute__((ext_vector_type(4))) unsigned int u32x4;
typedef __attribute__((ext_vector_type(8))) int         i32x8;
typedef __attribute__((ext_vector_type(4))) int         i32x4;

#define BATCH   8
#define SEQ     2048
#define DIM     64
#define RANK    32
#define NST     64
#define NPROJ   (RANK + 2 * NST)     // 160
#define CHUNK   32
#define NCHUNK  (SEQ / CHUNK)        // 64

static __device__ __forceinline__ bf16_t f2bf(float x) { return (bf16_t)x; }

static __device__ __forceinline__ unsigned int lds_off(const void* p) {
  // LDS flat addresses carry the LDS byte offset in the low 32 bits.
  return (unsigned int)(unsigned long long)p;
}

// Tensor Data Mover: contiguous 1-D copy of `elems` f32 from global to LDS.
// D# packing per CDNA5 ISA:
//   group0: [1:0]=count=1, [63:32]=lds_addr, [120:64]=global_addr, [127:126]=2
//   group1: [17:16]=data_size(4B), [79:48]=tensor_dim0, [127:112]=tile_dim0,
//           [207:160]=tensor_dim0_stride ; groups 2/3 = 0 (<=2-D tensor)
static __device__ __forceinline__ void tdm_load_1d(unsigned int lds_addr,
                                                   const void* gptr,
                                                   unsigned int elems) {
  unsigned long long ga = (unsigned long long)gptr;
  u32x4 g0;
  g0[0] = 1u;                                                  // count=1
  g0[1] = lds_addr;
  g0[2] = (unsigned int)ga;                                    // addr[31:0]
  g0[3] = (unsigned int)((ga >> 32) & 0x01FFFFFFu) | (2u << 30); // addr[56:32]|type=2
  i32x8 g1;
  g1[0] = (int)(2u << 16);                                     // data_size=4B
  g1[1] = (int)((elems & 0xFFFFu) << 16);                      // tensor_dim0[15:0]
  g1[2] = (int)(((elems >> 16) & 0xFFFFu) | (1u << 16));       // dim0[31:16], dim1=1
  g1[3] = (int)((elems & 0xFFFFu) << 16);                      // tile_dim0
  g1[4] = 0;                                                   // tile_dim1/2 unused
  g1[5] = (int)elems;                                          // dim0_stride[31:0]
  g1[6] = 0;
  g1[7] = 0;
  i32x4 gz = {0, 0, 0, 0};
#if __has_include(<hip/amd_detail/amd_gfx1250_TDM.h>)
  i32x8 gz8 = {0, 0, 0, 0, 0, 0, 0, 0};
  __builtin_amdgcn_tensor_load_to_lds(g0, g1, gz, gz, gz8, 0);
#else
  __builtin_amdgcn_tensor_load_to_lds(g0, g1, gz, gz, 0);
#endif
}

// A-fragment (16x32 bf16, M rows x K), source row-major in LDS.
static __device__ __forceinline__ v16bf load_a_frag(const bf16_t* p, int stride,
                                                    int row0, int k0) {
  const int lane = threadIdx.x & 31;
  const bf16_t* q = p + (row0 + (lane & 15)) * stride + k0 + ((lane >> 4) << 3);
  v8bf lo = *(const v8bf*)q;
  v8bf hi = *(const v8bf*)(q + 16);
  v16bf f;
#pragma unroll
  for (int i = 0; i < 8; ++i) { f[i] = lo[i]; f[i + 8] = hi[i]; }
  return f;
}

// B-fragment (32x16 bf16, K x N), source stored N-MAJOR (B^T row-major).
static __device__ __forceinline__ v16bf load_b_frag(const bf16_t* bt, int stride,
                                                    int n0, int k0) {
  const int lane = threadIdx.x & 31;
  const bf16_t* q = bt + (n0 + (lane & 15)) * stride + k0 + ((lane >> 4) << 3);
  v8bf lo = *(const v8bf*)q;
  v8bf hi = *(const v8bf*)(q + 16);
  v16bf f;
#pragma unroll
  for (int i = 0; i < 8; ++i) { f[i] = lo[i]; f[i + 8] = hi[i]; }
  return f;
}

static __device__ __forceinline__ v8f wmma_bf16(v16bf a, v16bf b, v8f c) {
  return __builtin_amdgcn_wmma_f32_16x16x32_bf16(false, a, false, b,
                                                 (short)0, c, false, false);
}

// ---------------------------------------------------------------------------
// Pass 0: u' = u + t_emb*scale ; x_proj = u' @ W_in + b_in ;
//         delta' = softplus(delta @ W_delta + b_delta).  One wave per 16 rows.
// ---------------------------------------------------------------------------
__global__ __launch_bounds__(32) void proj_kernel(
    const float* __restrict__ u, const float* __restrict__ t_emb,
    const float* __restrict__ W_in, const float* __restrict__ b_in,
    const float* __restrict__ W_delta, const float* __restrict__ b_delta,
    const float* __restrict__ scale,
    float* __restrict__ uP, float* __restrict__ dP,
    float* __restrict__ Bp, float* __restrict__ Cp) {
  __shared__ __align__(16) bf16_t u_lds[16 * DIM];
  __shared__ __align__(16) bf16_t wt_lds[NPROJ * DIM];    // W_in^T (n-major)
  __shared__ __align__(16) bf16_t wd_lds[DIM * RANK];     // W_delta^T (n-major)
  __shared__ __align__(16) bf16_t dl_lds[16 * RANK];      // delta tile

  const int tile = blockIdx.x;
  const int b  = tile / (SEQ / 16);
  const int l0 = (tile % (SEQ / 16)) * 16;
  const int lane = threadIdx.x;
  const float sc = scale[0];

  for (int i = lane; i < 16 * DIM; i += 32) {
    const int m = i >> 6, k = i & 63;
    const float v = u[((size_t)(b * SEQ + l0 + m)) * DIM + k] +
                    t_emb[b * DIM + k] * sc;
    uP[((size_t)(b * SEQ + l0 + m)) * DIM + k] = v;
    u_lds[i] = f2bf(v);
  }
  for (int i = lane; i < NPROJ * DIM; i += 32) {
    const int n = i >> 6, k = i & 63;
    wt_lds[i] = f2bf(W_in[k * NPROJ + n]);
  }
  for (int i = lane; i < DIM * RANK; i += 32) {
    const int n = i / RANK, k = i % RANK;
    wd_lds[i] = f2bf(W_delta[k * DIM + n]);
  }
  __syncthreads();

  const int hl = lane >> 4, nl = lane & 15;

  for (int jt = 0; jt < NPROJ / 16; ++jt) {
    v8f acc = {};
#pragma unroll
    for (int kk = 0; kk < DIM; kk += 32) {
      v16bf a  = load_a_frag(u_lds, DIM, 0, kk);
      v16bf bb = load_b_frag(wt_lds, DIM, jt * 16, kk);
      acc = wmma_bf16(a, bb, acc);
    }
    const float bias = b_in[jt * 16 + nl];
#pragma unroll
    for (int r = 0; r < 8; ++r) {
      const int m = r + 8 * hl;
      const int col = jt * 16 + nl;
      const float v = acc[r] + bias;
      if (col < RANK) {
        dl_lds[m * RANK + col] = f2bf(v);
      } else if (col < RANK + NST) {
        Bp[((size_t)(b * SEQ + l0 + m)) * NST + (col - RANK)] = v;
      } else {
        Cp[((size_t)(b * SEQ + l0 + m)) * NST + (col - RANK - NST)] = v;
      }
    }
  }
  __syncthreads();

  for (int jt = 0; jt < DIM / 16; ++jt) {
    v16bf a  = load_a_frag(dl_lds, RANK, 0, 0);
    v16bf bb = load_b_frag(wd_lds, RANK, jt * 16, 0);
    v8f acc = {};
    acc = wmma_bf16(a, bb, acc);
    const float bias = b_delta[jt * 16 + nl];
#pragma unroll
    for (int r = 0; r < 8; ++r) {
      const int m = r + 8 * hl;
      const float x = acc[r] + bias;
      const float sp = (x > 20.f) ? x : log1pf(__expf(x));
      dP[((size_t)(b * SEQ + l0 + m)) * DIM + jt * 16 + nl] = sp;
    }
  }
}

// ---------------------------------------------------------------------------
// Pass 1: per (batch, chunk): M_c = dA_{31}...dA_0  and  Zp_c = scan from 0.
// Chunk operands (A, delta', u', B') staged once via TDM; 32-step loop is
// LDS/WMMA-only.  16 waves, wave (ti,tj) owns a 16x16 tile; z/M ping-pong
// through LDS transposed so C/D frags re-enter as B operands.
// ---------------------------------------------------------------------------
__global__ __launch_bounds__(512) void chunk_kernel(
    const float* __restrict__ Amat, const float* __restrict__ dP,
    const float* __restrict__ uP, const float* __restrict__ Bp,
    float* __restrict__ Mc, float* __restrict__ Zp) {
  __shared__ __align__(16) float  A_f[DIM * NST];        // TDM
  __shared__ __align__(16) float  dPc[CHUNK * DIM];      // TDM
  __shared__ __align__(16) float  uPc[CHUNK * DIM];      // TDM
  __shared__ __align__(16) float  Bpc[CHUNK * NST];      // TDM
  __shared__ __align__(16) bf16_t dA[DIM * NST];
  __shared__ __align__(16) bf16_t zt[2][NST * DIM];
  __shared__ __align__(16) bf16_t mt[2][NST * DIM];

  const int b = blockIdx.x / NCHUNK;
  const int c = blockIdx.x % NCHUNK;
  const int tid  = threadIdx.x;
  const int wave = tid >> 5, lane = tid & 31;
  const int ti = wave >> 2, tj = wave & 3;
  const int hl = lane >> 4, nl = lane & 15;
  const size_t rowbase = (size_t)(b * SEQ + c * CHUNK);

  if (wave == 0) {
    tdm_load_1d(lds_off(A_f), Amat, DIM * NST);
    tdm_load_1d(lds_off(dPc), dP + rowbase * DIM, CHUNK * DIM);
    tdm_load_1d(lds_off(uPc), uP + rowbase * DIM, CHUNK * DIM);
    tdm_load_1d(lds_off(Bpc), Bp + rowbase * NST, CHUNK * NST);
    __builtin_amdgcn_s_wait_tensorcnt(0);
  }
  for (int i = tid; i < DIM * NST; i += 512) {
    zt[0][i] = f2bf(0.f);
    const int n = i >> 6, k = i & 63;
    mt[0][i] = f2bf(n == k ? 1.f : 0.f);                  // M = I
  }
  __syncthreads();

  v8f accz = {}, accm = {};
  int cur = 0;
  for (int j = 0; j < CHUNK; ++j) {
    for (int i = tid; i < DIM * NST; i += 512) {
      const int d = i >> 6;
      dA[i] = f2bf(__expf(dPc[j * DIM + d] * A_f[i]));
    }
    __syncthreads();

    const float bn = Bpc[j * NST + tj * 16 + nl];
#pragma unroll
    for (int r = 0; r < 8; ++r) {
      const int m = ti * 16 + r + 8 * hl;
      accz[r] = bn * uPc[j * DIM + m] * dPc[j * DIM + m];  // dBu seed
      accm[r] = 0.f;
    }
#pragma unroll
    for (int kk = 0; kk < NST; kk += 32) {
      v16bf a  = load_a_frag(dA, NST, ti * 16, kk);
      v16bf bz = load_b_frag(zt[cur], DIM, tj * 16, kk);
      v16bf bm = load_b_frag(mt[cur], DIM, tj * 16, kk);
      accz = wmma_bf16(a, bz, accz);
      accm = wmma_bf16(a, bm, accm);
    }
    const int nxt = cur ^ 1;
    v8bf pz, pm;
#pragma unroll
    for (int r = 0; r < 8; ++r) { pz[r] = f2bf(accz[r]); pm[r] = f2bf(accm[r]); }
    const int off = (tj * 16 + nl) * DIM + ti * 16 + 8 * hl;
    *(v8bf*)&zt[nxt][off] = pz;
    *(v8bf*)&mt[nxt][off] = pm;
    __syncthreads();
    cur = nxt;
  }

  const size_t base = ((size_t)(b * NCHUNK + c)) * DIM * NST;
#pragma unroll
  for (int r = 0; r < 8; ++r) {
    const int d = ti * 16 + r + 8 * hl, n = tj * 16 + nl;
    Zp[base + d * NST + n] = accz[r];
    Mc[base + d * NST + n] = accm[r];
  }
}

// ---------------------------------------------------------------------------
// Pass 2: per batch, sequential: zs_{c+1} = M_c @ zs_c + Zp_c.
// Next chunk's M_c is TDM-prefetched (double buffer) while the current one
// computes; s_wait_tensorcnt(1) retires the oldest in-order TDM op.
// ---------------------------------------------------------------------------
__global__ __launch_bounds__(512) void combine_kernel(
    const float* __restrict__ Mc, const float* __restrict__ Zp,
    float* __restrict__ Zs) {
  __shared__ __align__(16) float  Mf[2][DIM * NST];       // TDM double buffer
  __shared__ __align__(16) bf16_t mA[DIM * NST];
  __shared__ __align__(16) bf16_t zt[2][NST * DIM];

  const int b = blockIdx.x;
  const int tid  = threadIdx.x;
  const int wave = tid >> 5, lane = tid & 31;
  const int ti = wave >> 2, tj = wave & 3;
  const int hl = lane >> 4, nl = lane & 15;
  const size_t bbase = (size_t)b * NCHUNK * DIM * NST;

  if (wave == 0) tdm_load_1d(lds_off(Mf[0]), Mc + bbase, DIM * NST);
  for (int i = tid; i < NST * DIM; i += 512) zt[0][i] = f2bf(0.f);
  v8f acc = {};
  int cur = 0;

  for (int c = 0; c < NCHUNK; ++c) {
    const size_t base = bbase + (size_t)c * DIM * NST;
#pragma unroll
    for (int r = 0; r < 8; ++r) {
      const int d = ti * 16 + r + 8 * hl, n = tj * 16 + nl;
      Zs[base + d * NST + n] = acc[r];                    // state at chunk start
    }
    if (wave == 0) {
      if (c + 1 < NCHUNK) {
        tdm_load_1d(lds_off(Mf[(c + 1) & 1]), Mc + base + DIM * NST, DIM * NST);
        __builtin_amdgcn_s_wait_tensorcnt(1);             // oldest (c) done
      } else {
        __builtin_amdgcn_s_wait_tensorcnt(0);
      }
    }
    __syncthreads();
    for (int i = tid; i < DIM * NST; i += 512) mA[i] = f2bf(Mf[c & 1][i]);
    __syncthreads();

    v8f nz;
#pragma unroll
    for (int r = 0; r < 8; ++r) {
      const int d = ti * 16 + r + 8 * hl, n = tj * 16 + nl;
      nz[r] = Zp[base + d * NST + n];
    }
#pragma unroll
    for (int kk = 0; kk < NST; kk += 32) {
      v16bf a  = load_a_frag(mA, NST, ti * 16, kk);
      v16bf bz = load_b_frag(zt[cur], DIM, tj * 16, kk);
      nz = wmma_bf16(a, bz, nz);
    }
    const int nxt = cur ^ 1;
    v8bf p;
#pragma unroll
    for (int r = 0; r < 8; ++r) p[r] = f2bf(nz[r]);
    *(v8bf*)&zt[nxt][(tj * 16 + nl) * DIM + ti * 16 + 8 * hl] = p;
    acc = nz;
    __syncthreads();
    cur = nxt;
  }
}

// ---------------------------------------------------------------------------
// Pass 3: replay each chunk from Zs, emit v_l = sig(t)*(z_l @ C_l) + u'*D.
// ---------------------------------------------------------------------------
__global__ __launch_bounds__(512) void final_kernel(
    const float* __restrict__ Amat, const float* __restrict__ dP,
    const float* __restrict__ uP, const float* __restrict__ Bp,
    const float* __restrict__ Cp, const float* __restrict__ Zs,
    const float* __restrict__ t_emb, const float* __restrict__ Dvec,
    float* __restrict__ out) {
  __shared__ __align__(16) float  A_f[DIM * NST];        // TDM
  __shared__ __align__(16) float  Ztmp[DIM * NST];       // TDM (chunk start z)
  __shared__ __align__(16) float  dPc[CHUNK * DIM];      // TDM
  __shared__ __align__(16) float  uPc[CHUNK * DIM];      // TDM
  __shared__ __align__(16) float  Bpc[CHUNK * NST];      // TDM
  __shared__ __align__(16) float  Cpc[CHUNK * NST];      // TDM
  __shared__ __align__(16) bf16_t dA[DIM * NST];
  __shared__ __align__(16) bf16_t zt[2][NST * DIM];
  __shared__ float sig[DIM], dd[DIM], vacc[DIM];

  const int b = blockIdx.x / NCHUNK;
  const int c = blockIdx.x % NCHUNK;
  const int tid  = threadIdx.x;
  const int wave = tid >> 5, lane = tid & 31;
  const int ti = wave >> 2, tj = wave & 3;
  const int hl = lane >> 4, nl = lane & 15;
  const size_t rowbase = (size_t)(b * SEQ + c * CHUNK);
  const size_t zbase = ((size_t)(b * NCHUNK + c)) * DIM * NST;

  if (wave == 0) {
    tdm_load_1d(lds_off(A_f),  Amat, DIM * NST);
    tdm_load_1d(lds_off(Ztmp), Zs + zbase, DIM * NST);
    tdm_load_1d(lds_off(dPc),  dP + rowbase * DIM, CHUNK * DIM);
    tdm_load_1d(lds_off(uPc),  uP + rowbase * DIM, CHUNK * DIM);
    tdm_load_1d(lds_off(Bpc),  Bp + rowbase * NST, CHUNK * NST);
    tdm_load_1d(lds_off(Cpc),  Cp + rowbase * NST, CHUNK * NST);
    __builtin_amdgcn_s_wait_tensorcnt(0);
  }
  if (tid < DIM) {
    const float te = t_emb[b * DIM + tid];
    sig[tid] = 1.f / (1.f + __expf(-te));
    dd[tid]  = Dvec[tid];
  }
  __syncthreads();
  for (int i = tid; i < DIM * NST; i += 512) {
    const int d = i >> 6, n = i & 63;
    zt[0][n * DIM + d] = f2bf(Ztmp[i]);                   // stage z^T
  }
  __syncthreads();

  int cur = 0;
  for (int j = 0; j < CHUNK; ++j) {
    for (int i = tid; i < DIM * NST; i += 512) {
      const int d = i >> 6;
      dA[i] = f2bf(__expf(dPc[j * DIM + d] * A_f[i]));
    }
    if (tid < DIM) vacc[tid] = 0.f;
    __syncthreads();

    v8f accz;
    const float bn = Bpc[j * NST + tj * 16 + nl];
#pragma unroll
    for (int r = 0; r < 8; ++r) {
      const int m = ti * 16 + r + 8 * hl;
      accz[r] = bn * uPc[j * DIM + m] * dPc[j * DIM + m];
    }
#pragma unroll
    for (int kk = 0; kk < NST; kk += 32) {
      v16bf a  = load_a_frag(dA, NST, ti * 16, kk);
      v16bf bz = load_b_frag(zt[cur], DIM, tj * 16, kk);
      accz = wmma_bf16(a, bz, accz);
    }
    const int nxt = cur ^ 1;
    v8bf p;
#pragma unroll
    for (int r = 0; r < 8; ++r) p[r] = f2bf(accz[r]);
    *(v8bf*)&zt[nxt][(tj * 16 + nl) * DIM + ti * 16 + 8 * hl] = p;

    // v_l[d] partials: reduce z[d,n]*C[n] across the 16 lanes of a half-wave
    const float cn = Cpc[j * NST + tj * 16 + nl];
#pragma unroll
    for (int r = 0; r < 8; ++r) {
      float part = accz[r] * cn;
      part += __shfl_xor(part, 1, 32);
      part += __shfl_xor(part, 2, 32);
      part += __shfl_xor(part, 4, 32);
      part += __shfl_xor(part, 8, 32);
      if (nl == 0) atomicAdd(&vacc[ti * 16 + r + 8 * hl], part);
    }
    __syncthreads();
    if (tid < DIM) {
      out[(rowbase + j) * DIM + tid] = vacc[tid] * sig[tid] + uPc[j * DIM + tid] * dd[tid];
    }
    __syncthreads();
    cur = nxt;
  }
}

// ---------------------------------------------------------------------------
extern "C" void kernel_launch(void* const* d_in, const int* in_sizes, int n_in,
                              void* d_out, int out_size, void* d_ws, size_t ws_size,
                              hipStream_t stream) {
  (void)in_sizes; (void)n_in; (void)out_size; (void)ws_size;
  const float* u       = (const float*)d_in[0];
  const float* t_emb   = (const float*)d_in[1];
  const float* W_in    = (const float*)d_in[2];
  const float* b_in    = (const float*)d_in[3];
  const float* W_delta = (const float*)d_in[4];
  const float* b_delta = (const float*)d_in[5];
  const float* Amat    = (const float*)d_in[6];
  const float* Dvec    = (const float*)d_in[7];
  const float* scale   = (const float*)d_in[8];
  float* out = (float*)d_out;

  float* uP = (float*)d_ws;                                  // B*S*D
  float* dP = uP + (size_t)BATCH * SEQ * DIM;                // B*S*D
  float* Bp = dP + (size_t)BATCH * SEQ * DIM;                // B*S*N
  float* Cp = Bp + (size_t)BATCH * SEQ * NST;                // B*S*N
  float* Mc = Cp + (size_t)BATCH * SEQ * NST;                // B*NC*D*N
  float* Zp = Mc + (size_t)BATCH * NCHUNK * DIM * NST;       // B*NC*D*N
  float* Zs = Zp + (size_t)BATCH * NCHUNK * DIM * NST;       // B*NC*D*N

  proj_kernel<<<BATCH * SEQ / 16, 32, 0, stream>>>(
      u, t_emb, W_in, b_in, W_delta, b_delta, scale, uP, dP, Bp, Cp);
  chunk_kernel<<<BATCH * NCHUNK, 512, 0, stream>>>(Amat, dP, uP, Bp, Mc, Zp);
  combine_kernel<<<BATCH, 512, 0, stream>>>(Mc, Zp, Zs);
  final_kernel<<<BATCH * NCHUNK, 512, 0, stream>>>(
      Amat, dP, uP, Bp, Cp, Zs, t_emb, Dvec, out);
}
